// LSM_41609643163840
// MI455X (gfx1250) — matrix operand order repaired and checked by reading, not compiled, not run on hardware
//
#include <hip/hip_runtime.h>
#include <hip/hip_bf16.h>
#include <math.h>

typedef __attribute__((ext_vector_type(16))) __bf16 v16bf;
typedef __attribute__((ext_vector_type(8)))  __bf16 v8bf;
typedef __attribute__((ext_vector_type(8)))  float  v8f;

#define HH    128
#define DN    128
#define DV    128
#define DD    64
#define BLK   64
#define RBX   512
#define DIM   7168
#define RC    1536
#define SEQ   2048
#define NCHUNK 32

// ---------------------------------------------------------------------------
// WMMA helpers: D = A(16x32 bf16) x B(32x16 bf16) + C(16x16 f32), wave32.
// A-frag (ISA 16-bit A 16x32): lane L (m=L%16): elems[0..7]=K kb..kb+7,
// elems[8..15]=K kb+16..kb+23 with kb=(L/16)*8  -> two 16B LDS reads.
// B-frag (ISA 16-bit B 32x16): lane L (n=L%16): same K order as A-frag,
// elems[0..7]=K kb..kb+7, elems[8..15]=K kb+16..kb+23, kb=(L/16)*8.
// ---------------------------------------------------------------------------
__device__ __forceinline__ v8f wmma_bf16(v16bf a, v16bf b, v8f c) {
  return __builtin_amdgcn_wmma_f32_16x16x32_bf16(false, a, false, b,
                                                 (short)0, c, false, false);
}
__device__ __forceinline__ v16bf frag_a(const __bf16* row, int kk, int lh) {
  v8bf lo = *(const v8bf*)(row + kk * 32 + lh * 8);
  v8bf hi = *(const v8bf*)(row + kk * 32 + lh * 8 + 16);
  return __builtin_shufflevector(lo, hi, 0,1,2,3,4,5,6,7,8,9,10,11,12,13,14,15);
}
__device__ __forceinline__ v16bf frag_b(const __bf16* row, int kk, int lh) {
  v8bf lo = *(const v8bf*)(row + kk * 32 + lh * 8);
  v8bf hi = *(const v8bf*)(row + kk * 32 + lh * 8 + 16);
  return __builtin_shufflevector(lo, hi, 0,1,2,3,4,5,6,7,8,9,10,11,12,13,14,15);
}

// Async global->LDS copy (CDNA5 TDM-lite path, tracked by ASYNCcnt).
// GVS mode: uniform 64-bit base in SGPRs + per-lane 32-bit byte offset.
// Generic pointers to LDS carry the LDS byte offset in their low 32 bits
__device__ __forceinline__ void async_b128(const void* base_sgpr, uint32_t goff,
                                           uint32_t lds_off) {
  asm volatile("global_load_async_to_lds_b128 %0, %1, %2"
               :: "v"(lds_off), "v"(goff), "s"(base_sgpr)
               : "memory");
}
__device__ __forceinline__ void wait_async0() {
  asm volatile("s_wait_asynccnt 0" ::: "memory");
}

// ---------------------------------------------------------------------------
// Elementwise fp32 -> bf16 conversion (grid-stride)
// ---------------------------------------------------------------------------
__global__ void cvt_bf16(const float* __restrict__ in, __bf16* __restrict__ out,
                         size_t n) {
  size_t i = (size_t)blockIdx.x * blockDim.x + threadIdx.x;
  size_t stride = (size_t)gridDim.x * blockDim.x;
  for (; i < n; i += stride) out[i] = (__bf16)in[i];
}

// bx[s][0:512] fp32 -> bf16 (strip dt_b columns)
__global__ void extract_bxl(const float* __restrict__ bx, __bf16* __restrict__ out) {
  int s = blockIdx.x;
  for (int c = threadIdx.x; c < RBX; c += blockDim.x)
    out[(size_t)s * RBX + c] = (__bf16)bx[(size_t)s * (RBX + DD) + c];
}

// ---------------------------------------------------------------------------
// Tiled WMMA GEMM: C[M,N] = A[M,K] * W[N,K]^T, bf16 in, fp32 accum.
// Block = 256 threads (8 waves). Tile 128x128, BK=32, double-buffered LDS
// filled by async global->LDS b128 loads (overlapped with WMMA on the other
// buffer). Wave (wm,wn) = (w>>1, w&1) computes 32x64 via acc[2][4].
// ---------------------------------------------------------------------------
template <bool OUT_BF16>
__global__ void __launch_bounds__(256) gemm_wmma(const __bf16* __restrict__ A,
                                                 const __bf16* __restrict__ W,
                                                 void* __restrict__ Cout,
                                                 int M, int N, int K) {
  __shared__ __bf16 As[2][128][32];
  __shared__ __bf16 Bs[2][128][32];
  const int tid  = threadIdx.x;
  const int wave = tid >> 5, lane = tid & 31;
  const int lm   = lane & 15, lh = lane >> 4;
  const int wm   = wave >> 1, wn = wave & 1;
  const int m0   = blockIdx.y * 128, n0 = blockIdx.x * 128;
  (void)M;

  // staging assignment: thread t moves 32B of A and 32B of W per tile
  const int r    = tid >> 1;          // 0..127
  const int cb   = (tid & 1) * 16;    // 0 or 16 (bf16 elems)
  const int arow = m0 + r;
  const int wrow = (n0 + r < N) ? (n0 + r) : (N - 1);  // clamp for N%128 != 0

  v8f acc[2][4] = {};

  auto stage = [&](int buf, int k0) {
    const uint32_t la = (uint32_t)(uintptr_t)&As[buf][r][cb];
    const uint32_t lb = (uint32_t)(uintptr_t)&Bs[buf][r][cb];
    const uint32_t ga = (uint32_t)(((size_t)arow * K + k0 + cb) * sizeof(__bf16));
    const uint32_t gb = (uint32_t)(((size_t)wrow * K + k0 + cb) * sizeof(__bf16));
    async_b128(A, ga,      la);
    async_b128(A, ga + 16, la + 16);
    async_b128(W, gb,      lb);
    async_b128(W, gb + 16, lb + 16);
  };

  stage(0, 0);
  wait_async0();
  __syncthreads();

  for (int k0 = 0; k0 < K; k0 += 32) {
    const int  cur  = (k0 >> 5) & 1;
    const bool more = (k0 + 32) < K;
    if (more) stage(cur ^ 1, k0 + 32);   // overlap with WMMA below

    v16bf af[2], bf[4];
    for (int sm = 0; sm < 2; ++sm)
      af[sm] = frag_a(&As[cur][wm * 32 + sm * 16 + lm][0], 0, lh);
    for (int sn = 0; sn < 4; ++sn)
      bf[sn] = frag_b(&Bs[cur][wn * 64 + sn * 16 + lm][0], 0, lh);
    for (int sm = 0; sm < 2; ++sm)
      for (int sn = 0; sn < 4; ++sn)
        acc[sm][sn] = wmma_bf16(af[sm], bf[sn], acc[sm][sn]);

    if (more) wait_async0();
    __syncthreads();
  }

  // D layout: lane L, vgpr j -> row = 8*(L/16)+j, col = L%16
  for (int sm = 0; sm < 2; ++sm)
    for (int sn = 0; sn < 4; ++sn) {
      const int orow = m0 + wm * 32 + sm * 16 + lh * 8;
      const int ocol = n0 + wn * 64 + sn * 16 + lm;
      if (ocol < N) {
        for (int j = 0; j < 8; ++j) {
          size_t idx = (size_t)(orow + j) * N + ocol;
          if (OUT_BF16) ((__bf16*)Cout)[idx] = (__bf16)acc[sm][sn][j];
          else          ((float*)Cout)[idx]  = acc[sm][sn][j];
        }
      }
    }
}

// ---------------------------------------------------------------------------
// dt[s,h] = softplus( dot(dt_c[s,h,:64], dt_b[s,:64]) )
// ---------------------------------------------------------------------------
__global__ void dt_softplus(const __bf16* __restrict__ cbuf,
                            const float* __restrict__ bx,
                            float* __restrict__ dtg) {
  const int s = blockIdx.x, h = threadIdx.x;
  __shared__ float db[DD];
  if (h < DD) db[h] = bx[(size_t)s * (RBX + DD) + RBX + h];
  __syncthreads();
  const __bf16* dc = cbuf + ((size_t)s * HH + h) * (DN + DD) + DN;
  float a = 0.f;
  for (int d = 0; d < DD; ++d) a += (float)dc[d] * db[d];
  dtg[(size_t)s * HH + h] = (a > 20.f) ? a : log1pf(expf(a));
}

// ---------------------------------------------------------------------------
// SSD chunked scan. Grid = (DV/64 p-halves, HH heads), 256 threads (8 waves).
// Running state prev[n=128][p=64] lives in WMMA accumulators (wave w owns
// n-tile w, p-tiles 0..3). Per chunk:
//   scores = (C B^T) * Lmat                   -> WMMA + elementwise decay/mask
//   Y      = scores*Xd + exp(Acs)*(C prev)    -> WMMA
//   prev   = exp(sum dA)*prev + (B*decay)^T Xd -> WMMA with scaled C operand
// ---------------------------------------------------------------------------
__global__ void __launch_bounds__(256) ssd_scan(
    const __bf16* __restrict__ cbuf,   // [S][H][DN+DD]
    const __bf16* __restrict__ bxf,    // [S][H][DN+DV]
    const float*  __restrict__ dtg,    // [S][H]
    const float*  __restrict__ A_log,  // [H]
    float* __restrict__ out) {         // [S][H][DV]
  __shared__ __bf16 Cb[BLK][DN];   // C chunk (l,n)
  __shared__ __bf16 Bb[BLK][DN];   // B chunk (s,n)
  __shared__ __bf16 Bt[DN][BLK];   // (B * decay_states)^T   (n,l)
  __shared__ __bf16 Xt[64][BLK];   // (X * dt)^T, p-half     (p,l)
  __shared__ __bf16 Sb[BLK][BLK];  // masked scores          (l,s)
  __shared__ __bf16 Pt[64][DN];    // prev state^T           (p,n)
  __shared__ float  Acs[BLK];
  __shared__ float  dtl[BLK];

  const int tid  = threadIdx.x;
  const int wave = tid >> 5, lane = tid & 31;
  const int lm   = lane & 15, lh = lane >> 4;
  const int p0   = blockIdx.x * 64;
  const int h    = blockIdx.y;
  const float Ah = -__expf(A_log[h]);

  v8f accS[4] = {};
  for (int i = tid; i < 64 * DN; i += 256) ((__bf16*)Pt)[i] = (__bf16)0.f;
  __syncthreads();

  for (int chk = 0; chk < NCHUNK; ++chk) {
    const int s0 = chk * BLK;
    if (tid < BLK) dtl[tid] = dtg[(size_t)(s0 + tid) * HH + h];
    __syncthreads();
    if (tid == 0) {
      float a = 0.f;
      for (int l = 0; l < BLK; ++l) { a += dtl[l] * Ah; Acs[l] = a; }
    }
    __syncthreads();
    const float csum = Acs[BLK - 1];

    // fill LDS tiles
    for (int i = tid; i < BLK * DN; i += 256) {
      const int l = i >> 7, n = i & (DN - 1);
      const size_t base = (size_t)(s0 + l) * HH + h;
      Cb[l][n] = cbuf[base * (DN + DD) + n];
      const float bv = (float)bxf[base * (DN + DV) + n];
      Bb[l][n] = (__bf16)bv;
      Bt[n][l] = (__bf16)(bv * __expf(csum - Acs[l]));
    }
    for (int i = tid; i < 64 * BLK; i += 256) {
      const int p = i >> 6, l = i & 63;
      const float xv = (float)bxf[((size_t)(s0 + l) * HH + h) * (DN + DV) + DN + p0 + p];
      Xt[p][l] = (__bf16)(xv * dtl[l]);
    }
    __syncthreads();

    // scores (64x64, K=128), 16 tiles / 8 waves
    for (int t = wave * 2; t <= wave * 2 + 1; ++t) {
      const int tm = t >> 2, tn = t & 3;
      v8f acc = {};
      for (int kk = 0; kk < 4; ++kk) {
        v16bf a = frag_a(&Cb[tm * 16 + lm][0], kk, lh);
        v16bf b = frag_b(&Bb[tn * 16 + lm][0], kk, lh);
        acc = wmma_bf16(a, b, acc);
      }
      const int sc = tn * 16 + lm;
      const float acs_s = Acs[sc];
      for (int j = 0; j < 8; ++j) {
        const int lr = tm * 16 + lh * 8 + j;
        const float v = (lr >= sc) ? acc[j] * __expf(Acs[lr] - acs_s) : 0.f;
        Sb[lr][sc] = (__bf16)v;
      }
    }
    __syncthreads();

    // Y = Sb @ Xd (K=64)  +  exp(Acs) * (Cb @ prev) (K=128)
    for (int t = wave * 2; t <= wave * 2 + 1; ++t) {
      const int tm = t >> 2, tp = t & 3;
      v8f accd = {};
      for (int kk = 0; kk < 2; ++kk) {
        v16bf a = frag_a(&Sb[tm * 16 + lm][0], kk, lh);
        v16bf b = frag_b(&Xt[tp * 16 + lm][0], kk, lh);
        accd = wmma_bf16(a, b, accd);
      }
      v8f acco = {};
      for (int kk = 0; kk < 4; ++kk) {
        v16bf a = frag_a(&Cb[tm * 16 + lm][0], kk, lh);
        v16bf b = frag_b(&Pt[tp * 16 + lm][0], kk, lh);
        acco = wmma_bf16(a, b, acco);
      }
      const int pc = tp * 16 + lm;
      for (int j = 0; j < 8; ++j) {
        const int lr = tm * 16 + lh * 8 + j;
        out[((size_t)(s0 + lr) * HH + h) * DV + p0 + pc] =
            accd[j] + __expf(Acs[lr]) * acco[j];
      }
    }

    // state update in registers: accS = lam*accS + Bt @ Xd (K=64)
    const float lam = __expf(csum);
    for (int tp = 0; tp < 4; ++tp) {
      v8f sc_;
      for (int j = 0; j < 8; ++j) sc_[j] = accS[tp][j] * lam;
      for (int kk = 0; kk < 2; ++kk) {
        v16bf a = frag_a(&Bt[wave * 16 + lm][0], kk, lh);
        v16bf b = frag_b(&Xt[tp * 16 + lm][0], kk, lh);
        sc_ = wmma_bf16(a, b, sc_);
      }
      accS[tp] = sc_;
    }
    __syncthreads();  // everyone done reading Pt before overwrite
    for (int tp = 0; tp < 4; ++tp) {
      const int pc = tp * 16 + lm;
      for (int j = 0; j < 8; ++j)
        Pt[pc][wave * 16 + lh * 8 + j] = (__bf16)accS[tp][j];
    }
    __syncthreads();
  }
}

// ---------------------------------------------------------------------------
extern "C" void kernel_launch(void* const* d_in, const int* in_sizes, int n_in,
                              void* d_out, int out_size, void* d_ws, size_t ws_size,
                              hipStream_t stream) {
  (void)in_sizes; (void)n_in; (void)out_size; (void)ws_size;
  const float* x     = (const float*)d_in[0];
  const float* wc_a  = (const float*)d_in[1];
  const float* wc_b  = (const float*)d_in[2];
  const float* wbx_a = (const float*)d_in[3];
  const float* wbx_b = (const float*)d_in[4];
  const float* A_log = (const float*)d_in[5];
  float* out = (float*)d_out;

  char* p = (char*)d_ws;
  auto carve = [&](size_t bytes) {
    void* r = (void*)p;
    p += (bytes + 255) & ~(size_t)255;
    return r;
  };
  __bf16* xb    = (__bf16*)carve((size_t)SEQ * DIM * 2);
  __bf16* wcab  = (__bf16*)carve((size_t)RC * DIM * 2);
  __bf16* wcbb  = (__bf16*)carve((size_t)HH * (DN + DD) * RC * 2);
  __bf16* wbxab = (__bf16*)carve((size_t)(RBX + DD) * DIM * 2);
  __bf16* wbxbb = (__bf16*)carve((size_t)HH * (DN + DV) * RBX * 2);
  __bf16* t1b   = (__bf16*)carve((size_t)SEQ * RC * 2);
  __bf16* cB    = (__bf16*)carve((size_t)SEQ * HH * (DN + DD) * 2);
  float*  bx    = (float*) carve((size_t)SEQ * (RBX + DD) * 4);
  __bf16* bxlb  = (__bf16*)carve((size_t)SEQ * RBX * 2);
  __bf16* bxfB  = (__bf16*)carve((size_t)SEQ * HH * (DN + DV) * 2);
  float*  dtg   = (float*) carve((size_t)SEQ * HH * 4);

  // bf16 conversions of inputs
  cvt_bf16<<<2048, 256, 0, stream>>>(x,     xb,    (size_t)SEQ * DIM);
  cvt_bf16<<<2048, 256, 0, stream>>>(wc_a,  wcab,  (size_t)RC * DIM);
  cvt_bf16<<<4096, 256, 0, stream>>>(wc_b,  wcbb,  (size_t)HH * (DN + DD) * RC);
  cvt_bf16<<<1024, 256, 0, stream>>>(wbx_a, wbxab, (size_t)(RBX + DD) * DIM);
  cvt_bf16<<<2048, 256, 0, stream>>>(wbx_b, wbxbb, (size_t)HH * (DN + DV) * RBX);

  // projection GEMM chain (bf16 WMMA, fp32 accum), 128x128 tiles, async LDS
  gemm_wmma<true ><<<dim3(RC / 128, SEQ / 128), 256, 0, stream>>>(xb, wcab, t1b, SEQ, RC, DIM);
  gemm_wmma<true ><<<dim3(HH * (DN + DD) / 128, SEQ / 128), 256, 0, stream>>>(t1b, wcbb, cB, SEQ, HH * (DN + DD), RC);
  gemm_wmma<false><<<dim3((RBX + DD + 127) / 128, SEQ / 128), 256, 0, stream>>>(xb, wbxab, bx, SEQ, RBX + DD, DIM);
  extract_bxl<<<SEQ, 256, 0, stream>>>(bx, bxlb);
  gemm_wmma<true ><<<dim3(HH * (DN + DV) / 128, SEQ / 128), 256, 0, stream>>>(bxlb, wbxbb, bxfB, SEQ, HH * (DN + DV), RBX);

  // dt + scan
  dt_softplus<<<SEQ, HH, 0, stream>>>(cB, bx, dtg);
  ssd_scan<<<dim3(DV / 64, HH), 256, 0, stream>>>(cB, bxfB, dtg, A_log, out);
}
